// MultiHeadRevGatingLayer_5987184410675
// MI455X (gfx1250) — matrix operand order repaired and checked by reading, not compiled, not run on hardware
//
#include <hip/hip_runtime.h>
#include <hip/hip_bf16.h>

// ---- problem constants (match reference) ----
#define N_NODES   50000
#define N_EDGES   800000
#define HIDDEN    128
#define STATIC_F  16
#define EDGE_F    16
#define HEADS     8
#define MLP_IN    304      // 2*128 + 2*16 + 16
#define MLP_INP   320      // padded to 10 K-chunks of 32
#define MLP_W     256      // hidden width (8 K-chunks of 32)

// LDS row strides (in bf16 elements), padded for bank-conflict-free
// ds_load_b128 across 16 rows:
//  STR1 = 328 halves = 164 dwords; 164 mod 64 = 36 -> 16 disjoint 4-dword spans
//  STR2 = 272 halves = 136 dwords; 136 mod 64 =  8 -> 8 disjoint 4-dword spans
#define STR1      328
#define STR2      272

typedef __attribute__((ext_vector_type(16))) __bf16 v16bf;
typedef __attribute__((ext_vector_type(8)))  __bf16 v8bf;
typedef __attribute__((ext_vector_type(4)))  __bf16 v4bf;
typedef __attribute__((ext_vector_type(8)))  float  v8f;

// ---------------------------------------------------------------------------
// Fragment loaders (layouts per CDNA5 ISA 7.12.2, wave32)
// A (16-bit, 16x32): lane: row = lane&15, kb = (lane>=16)*8
//   elems 0..7  = K kb..kb+7       (at p+0)
//   elems 8..15 = K kb+16..kb+23   (at p+16)
// B (16-bit, 32x16): lane: col = lane&15, same K split as A (kb..kb+7, kb+16..kb+23)
// ---------------------------------------------------------------------------
template <typename P>
__device__ inline v16bf ldA_frag(P p) {
  v8bf lo = *(const v8bf*)(p);
  v8bf hi = *(const v8bf*)(p + 16);
  v16bf r;
#pragma unroll
  for (int i = 0; i < 8; ++i) { r[i] = lo[i]; r[i + 8] = hi[i]; }
  return r;
}

template <typename P>
__device__ inline v16bf ldB_frag(P p) {
  v8bf lo = *(const v8bf*)(p);
  v8bf hi = *(const v8bf*)(p + 16);
  v16bf r;
#pragma unroll
  for (int i = 0; i < 8; ++i) { r[i] = lo[i]; r[i + 8] = hi[i]; }
  return r;
}

__device__ inline v8f wmma_bf16(v16bf a, v16bf b, v8f c) {
  return __builtin_amdgcn_wmma_f32_16x16x32_bf16(
      /*neg_a=*/false, a, /*neg_b=*/false, b,
      /*c_mod=*/(short)0, c, /*reuse_a=*/false, /*reuse_b=*/false);
}

__device__ inline v8f zero_v8f() {
  v8f z;
#pragma unroll
  for (int i = 0; i < 8; ++i) z[i] = 0.f;
  return z;
}

// ---------------------------------------------------------------------------
// Kernel 0: zero the fp32 aggregation buffer (atomics accumulate into it)
// ---------------------------------------------------------------------------
__global__ __launch_bounds__(256) void k_zero(float4* __restrict__ agg, int n4) {
  int i = blockIdx.x * blockDim.x + threadIdx.x;
  int stride = gridDim.x * blockDim.x;
  float4 z; z.x = z.y = z.z = z.w = 0.f;
  for (; i < n4; i += stride) agg[i] = z;
}

// ---------------------------------------------------------------------------
// Kernel 1: convert weights fp32 -> bf16, N-major (row n holds K-contiguous)
//   W1b: [256][320]  (K padded, zeros in 304..319)
//   W2b: [16][256]   (N padded 8->16, zeros in cols 8..15)
//   Wob: [128][128]
// ---------------------------------------------------------------------------
__global__ __launch_bounds__(256) void k_convert(
    const float* __restrict__ W1, const float* __restrict__ W2,
    const float* __restrict__ Wout, __bf16* __restrict__ W1b,
    __bf16* __restrict__ W2b, __bf16* __restrict__ Wob) {
  const int TOT1 = MLP_W * MLP_INP;     // 81920
  const int TOT2 = 16 * MLP_W;          // 4096
  const int TOT3 = HIDDEN * HIDDEN;     // 16384
  int i = blockIdx.x * blockDim.x + threadIdx.x;
  int stride = gridDim.x * blockDim.x;
  for (; i < TOT1 + TOT2 + TOT3; i += stride) {
    if (i < TOT1) {
      int n = i / MLP_INP, k = i - n * MLP_INP;
      W1b[i] = (k < MLP_IN) ? (__bf16)W1[n * MLP_IN + k] : (__bf16)0.f;
    } else if (i < TOT1 + TOT2) {
      int j = i - TOT1;
      int n = j / MLP_W, k = j - n * MLP_W;
      W2b[j] = (n < HEADS) ? (__bf16)W2[n * MLP_W + k] : (__bf16)0.f;
    } else {
      int j = i - TOT1 - TOT2;
      Wob[j] = (__bf16)Wout[j];
    }
  }
}

// ---------------------------------------------------------------------------
// Kernel 2: per-edge gate MLP (WMMA bf16) + gather-multiply-scatter
// One workgroup = 64 edges, 256 threads (8 waves).
// ---------------------------------------------------------------------------
__global__ __launch_bounds__(256) void k_edges(
    const float* __restrict__ h, const float* __restrict__ xs,
    const int* __restrict__ ei, const float* __restrict__ ef,
    const __bf16* __restrict__ W1b, const __bf16* __restrict__ W2b,
    float* __restrict__ agg, float* __restrict__ gates_out) {
  // LDS union buffer: mlp_input [64][STR1] bf16 (~41KB), later hidden [64][STR2]
  __shared__ __bf16 buf[64 * STR1];
  __shared__ int   sSrc[64], sDst[64];
  __shared__ float sGates[64 * HEADS];

  const int t    = threadIdx.x;
  const int lane = t & 31;
  const int wave = t >> 5;
  const int hi   = lane >> 4;    // half-wave select
  const int lm   = lane & 15;
  const int e0   = blockIdx.x * 64;

  if (t < 64) {
    sDst[t] = ei[e0 + t];            // edge_index[0] = dest
    sSrc[t] = ei[N_EDGES + e0 + t];  // edge_index[1] = src
  }
  __syncthreads();

  // ---- Phase 1: gather + concat into LDS as bf16, float4-vectorized ----
  // 64 edges x 80 float4-groups; all segment boundaries are multiples of 4.
  for (int g = t; g < 64 * 80; g += 256) {
    int e  = g / 80;
    int k4 = (g - e * 80) * 4;
    int s = sSrc[e], d = sDst[e];
    float4 v; v.x = v.y = v.z = v.w = 0.f;
    if (k4 < 128)      v = *(const float4*)(h  + s * HIDDEN   + k4);
    else if (k4 < 256) v = *(const float4*)(h  + d * HIDDEN   + (k4 - 128));
    else if (k4 < 272) v = *(const float4*)(xs + s * STATIC_F + (k4 - 256));
    else if (k4 < 288) v = *(const float4*)(xs + d * STATIC_F + (k4 - 272));
    else if (k4 < 304) v = *(const float4*)(ef + (e0 + e) * EDGE_F + (k4 - 288));
    v4bf b;
    b[0] = (__bf16)v.x; b[1] = (__bf16)v.y;
    b[2] = (__bf16)v.z; b[3] = (__bf16)v.w;
    *(v4bf*)&buf[e * STR1 + k4] = b;     // 8B-aligned ds_store_b64
  }
  __syncthreads();

  // ---- Phase 2: GEMM1  [64,320]bf16 @ [320,256]bf16 -> fp32 acc ----
  // wave owns N columns [wave*32, wave*32+32); 4 M-tiles of 16 edges
  const int n0 = wave * 32;
  v8f acc[4][2];
#pragma unroll
  for (int mt = 0; mt < 4; ++mt) { acc[mt][0] = zero_v8f(); acc[mt][1] = zero_v8f(); }

#pragma unroll
  for (int kc = 0; kc < 10; ++kc) {
    v16bf b0 = ldB_frag(W1b + (n0      + lm) * MLP_INP + kc * 32 + hi * 8);
    v16bf b1 = ldB_frag(W1b + (n0 + 16 + lm) * MLP_INP + kc * 32 + hi * 8);
#pragma unroll
    for (int mt = 0; mt < 4; ++mt) {
      v16bf a = ldA_frag(&buf[(mt * 16 + lm) * STR1 + kc * 32 + hi * 8]);
      acc[mt][0] = wmma_bf16(a, b0, acc[mt][0]);
      acc[mt][1] = wmma_bf16(a, b1, acc[mt][1]);
    }
  }
  __syncthreads();   // everyone finished reading mlp_input

  // ---- Phase 3: relu + bf16, store hidden [64][STR2] into same LDS buffer ----
#pragma unroll
  for (int mt = 0; mt < 4; ++mt)
#pragma unroll
    for (int nt = 0; nt < 2; ++nt)
#pragma unroll
      for (int r = 0; r < 8; ++r) {
        int row = mt * 16 + r + 8 * hi;          // C/D layout
        int col = n0 + nt * 16 + lm;
        buf[row * STR2 + col] = (__bf16)fmaxf(acc[mt][nt][r], 0.f);
      }
  __syncthreads();

  // ---- Phase 4: GEMM2 [64,256] @ [256,16(8 valid)] -> sigmoid gates ----
  if (wave < 4) {
    const int mt = wave;   // 16 edges per wave
    v8f g = zero_v8f();
#pragma unroll
    for (int kc = 0; kc < 8; ++kc) {
      v16bf a = ldA_frag(&buf[(mt * 16 + lm) * STR2 + kc * 32 + hi * 8]);
      v16bf b = ldB_frag(W2b + lm * MLP_W + kc * 32 + hi * 8);
      g = wmma_bf16(a, b, g);
    }
    if (lm < HEADS) {
#pragma unroll
      for (int r = 0; r < 8; ++r) {
        int e = mt * 16 + r + 8 * hi;
        float gate = 1.f / (1.f + __expf(-g[r]));
        sGates[e * HEADS + lm] = gate;
        gates_out[(e0 + e) * HEADS + lm] = gate;   // output #2
      }
    }
  }
  __syncthreads();

  // ---- Phase 5: gated scatter-add:  agg[dest] += h[src] * gate[head] ----
  // 4 threads per edge; each covers 32 contiguous features (2 heads)
  {
    int e    = t & 63;
    int part = t >> 6;                 // 0..3
    int s = sSrc[e], d = sDst[e];
    float g0 = sGates[e * HEADS + part * 2];
    float g1 = sGates[e * HEADS + part * 2 + 1];
    const float4* hp4 = (const float4*)(h + s * HIDDEN + part * 32);
    float*        ap  = agg + d * HIDDEN + part * 32;
#pragma unroll
    for (int q = 0; q < 8; ++q) {
      float4 x = hp4[q];
      float gq = (q < 4) ? g0 : g1;
      atomicAdd(&ap[q * 4 + 0], x.x * gq);
      atomicAdd(&ap[q * 4 + 1], x.y * gq);
      atomicAdd(&ap[q * 4 + 2], x.z * gq);
      atomicAdd(&ap[q * 4 + 3], x.w * gq);
    }
  }
}

// ---------------------------------------------------------------------------
// Kernel 3: projection  out = agg @ Wout.T   ([50000,128] @ [128,128])
// One workgroup per 16-node tile; 8 waves x 1 N-tile; agg -> bf16 on the fly.
// ---------------------------------------------------------------------------
__global__ __launch_bounds__(256) void k_proj(
    const float* __restrict__ agg, const __bf16* __restrict__ Wob,
    float* __restrict__ out) {
  const int t = threadIdx.x, lane = t & 31, wave = t >> 5;
  const int hi = lane >> 4, lm = lane & 15;
  const int r0 = blockIdx.x * 16;

  v8f acc = zero_v8f();
#pragma unroll
  for (int kc = 0; kc < 4; ++kc) {
    // A fragment: row = r0+lm, K = kc*32 + kb + {0..7, 16..23}, fp32 -> bf16
    const float* ap = agg + (r0 + lm) * HIDDEN + kc * 32 + hi * 8;
    float4 q0 = ((const float4*)ap)[0];
    float4 q1 = ((const float4*)ap)[1];
    float4 q2 = ((const float4*)ap)[4];   // +16 floats
    float4 q3 = ((const float4*)ap)[5];
    v16bf a;
    a[0]=(__bf16)q0.x; a[1]=(__bf16)q0.y; a[2]=(__bf16)q0.z; a[3]=(__bf16)q0.w;
    a[4]=(__bf16)q1.x; a[5]=(__bf16)q1.y; a[6]=(__bf16)q1.z; a[7]=(__bf16)q1.w;
    a[8]=(__bf16)q2.x; a[9]=(__bf16)q2.y; a[10]=(__bf16)q2.z; a[11]=(__bf16)q2.w;
    a[12]=(__bf16)q3.x; a[13]=(__bf16)q3.y; a[14]=(__bf16)q3.z; a[15]=(__bf16)q3.w;
    v16bf b = ldB_frag(Wob + (wave * 16 + lm) * HIDDEN + kc * 32 + hi * 8);
    acc = wmma_bf16(a, b, acc);
  }
#pragma unroll
  for (int r = 0; r < 8; ++r)
    out[(r0 + r + 8 * hi) * HIDDEN + wave * 16 + lm] = acc[r];
}

// ---------------------------------------------------------------------------
extern "C" void kernel_launch(void* const* d_in, const int* in_sizes, int n_in,
                              void* d_out, int out_size, void* d_ws, size_t ws_size,
                              hipStream_t stream) {
  const float* h    = (const float*)d_in[0];
  const float* xs   = (const float*)d_in[1];
  const int*   ei   = (const int*)  d_in[2];
  const float* ef   = (const float*)d_in[3];
  const float* W1   = (const float*)d_in[4];
  const float* W2   = (const float*)d_in[5];
  const float* Wout = (const float*)d_in[6];

  float* out_proj  = (float*)d_out;                            // [N_NODES,128]
  float* out_gates = (float*)d_out + (size_t)N_NODES * HIDDEN; // [N_EDGES,8]

  // workspace layout (all offsets 256B aligned)
  char* ws = (char*)d_ws;
  float*  agg = (float*)ws;                                   // 25.6 MB
  __bf16* W1b = (__bf16*)(ws + (size_t)N_NODES * HIDDEN * 4); // 160 KB
  __bf16* W2b = (__bf16*)((char*)W1b + (size_t)MLP_W * MLP_INP * 2); // 8 KB
  __bf16* Wob = (__bf16*)((char*)W2b + (size_t)16 * MLP_W * 2);      // 32 KB

  k_zero<<<1600, 256, 0, stream>>>((float4*)agg, N_NODES * HIDDEN / 4);
  k_convert<<<400, 256, 0, stream>>>(W1, W2, Wout, W1b, W2b, Wob);
  k_edges<<<N_EDGES / 64, 256, 0, stream>>>(h, xs, ei, ef, W1b, W2b, agg, out_gates);
  k_proj<<<N_NODES / 16, 256, 0, stream>>>(agg, Wob, out_proj);
}
